// Transformer_84636625535040
// MI455X (gfx1250) — matrix operand order repaired and checked by reading, not compiled, not run on hardware
//
#include <hip/hip_runtime.h>
#include <hip/hip_bf16.h>

// ---------------------------------------------------------------------------
// Transformer encoder forward (L=4, S=512, B=8, D=1024, H=16, DFF=4096)
// All GEMMs run on CDNA5 WMMA: v_wmma_f32_16x16x32_bf16 (fp32 -> bf16 via
// v_cvt_pk_bf16_f32, fp32 accumulate).  wave32, 8 waves/block, 128x64 block
// tile, K-step 64, double-buffered LDS with register prefetch so global
// loads overlap the 8 WMMAs of the previous tile.
// ---------------------------------------------------------------------------

#define L_  4
#define S_  512
#define B_  8
#define D_  1024
#define H_  16
#define DFF_ 4096
#define HD_ 64
#define SB_ (S_*B_)            // 4096 tokens
#define BH_ (B_*H_)            // 128 attention batches
#define XE_ ((long long)SB_*D_)        // elems of x in d_out
#define SS2_ ((long long)S_*S_)        // 262144

typedef __bf16 bf16_t;
typedef bf16_t v16bf __attribute__((ext_vector_type(16)));
typedef float  v8f   __attribute__((ext_vector_type(8)));

// Packed fp32x2 -> bf16x2 (RNE) in one VOP3: dst.lo = cvt(a), dst.hi = cvt(b)
__device__ __forceinline__ unsigned int pack2bf(float a, float b) {
  unsigned int r;
  asm("v_cvt_pk_bf16_f32 %0, %1, %2" : "=v"(r) : "v"(a), "v"(b));
  return r;
}

// ---------------------------------------------------------------------------
// Generic batched GEMM:  C[z][m][n] = scale * sum_k A[z][m][k]*B'[z][k][n] + bias[n]
//   btrans==0: B given as [N][K] row-major (k contiguous)  -> C = A * B^T
//   btrans==1: B given as [K][N] row-major (n contiguous)  -> C = A * B
// Per-batch offsets: off = (z/zdiv)*S1 + (z%zdiv)*S0  for A, B, C.
// Requires M%128==0, N%64==0, K%64==0 (true for every GEMM in this model).
// ---------------------------------------------------------------------------
__global__ __launch_bounds__(256) void gemm_bf16_wmma(
    const float* __restrict__ A, long long aS1, long long aS0,
    const float* __restrict__ Bm, long long bS1, long long bS0,
    const float* __restrict__ bias,
    float* __restrict__ C, long long cS1, long long cS0,
    int zdiv, int lda, int ldb, int ldc, int K,
    float scale, int relu, int btrans)
{
  __shared__ unsigned short As[2][128 * 72];   // 128 rows x 64 k, pad 72, x2 buf
  __shared__ unsigned short Bs[2][ 64 * 72];   //  64 cols x 64 k, pad 72, x2 buf

  const int z = blockIdx.z;
  const float* a = A  + (long long)(z / zdiv) * aS1 + (long long)(z % zdiv) * aS0;
  const float* b = Bm + (long long)(z / zdiv) * bS1 + (long long)(z % zdiv) * bS0;
  float*       c = C  + (long long)(z / zdiv) * cS1 + (long long)(z % zdiv) * cS0;

  const int tid  = threadIdx.x;
  const int lane = tid & 31;
  const int wave = tid >> 5;
  const int wm   = wave & 3;        // 4 wave-rows  -> 128 M
  const int wn   = wave >> 2;       // 2 wave-cols  ->  64 N
  const int half = lane >> 4;
  const int l16  = lane & 15;
  const int seg  = tid & 15;        // 16 segs of 4 floats = 64 wide
  const int row0 = tid >> 4;        // 0..15
  const long long mBase = (long long)blockIdx.y * 128;
  const long long nBase = (long long)blockIdx.x * 64;

  v8f acc[2][2] = {};
  union Frag { uint4 q[2]; v16bf v; };

  uint2 aR[8], bR[4];               // register-staged bf16 tile (prefetch)

  // ---- load tile at k0 into registers (global -> cvt -> regs) ----
  auto globalLoad = [&](int k0) {
    #pragma unroll
    for (int p = 0; p < 8; ++p) {
      const int r = row0 + p * 16;
      const float4 f = *(const float4*)(a + (mBase + r) * (long long)lda + k0 + seg * 4);
      aR[p].x = pack2bf(f.x, f.y);
      aR[p].y = pack2bf(f.z, f.w);
    }
    if (btrans == 0) {
      #pragma unroll
      for (int p = 0; p < 4; ++p) {
        const int n = row0 + p * 16;
        const float4 f = *(const float4*)(b + (nBase + n) * (long long)ldb + k0 + seg * 4);
        bR[p].x = pack2bf(f.x, f.y);
        bR[p].y = pack2bf(f.z, f.w);
      }
    } else {
      #pragma unroll
      for (int p = 0; p < 4; ++p) {
        const int k = row0 + p * 16;
        const float4 f = *(const float4*)(b + (long long)(k0 + k) * ldb + nBase + seg * 4);
        bR[p].x = pack2bf(f.x, f.y);
        bR[p].y = pack2bf(f.z, f.w);
      }
    }
  };

  // ---- store register-staged tile into LDS buffer `buf` ----
  auto stageStore = [&](int buf) {
    unsigned short* Asb = &As[buf][0];
    unsigned short* Bsb = &Bs[buf][0];
    #pragma unroll
    for (int p = 0; p < 8; ++p) {
      unsigned int* dst = (unsigned int*)&Asb[(row0 + p * 16) * 72 + seg * 4];
      dst[0] = aR[p].x;
      dst[1] = aR[p].y;
    }
    if (btrans == 0) {
      #pragma unroll
      for (int p = 0; p < 4; ++p) {
        unsigned int* dst = (unsigned int*)&Bsb[(row0 + p * 16) * 72 + seg * 4];
        dst[0] = bR[p].x;
        dst[1] = bR[p].y;
      }
    } else {                         // transpose in LDS: Bs[n][k]
      #pragma unroll
      for (int p = 0; p < 4; ++p) {
        const int k = row0 + p * 16;
        const int n = seg * 4;
        Bsb[(n + 0) * 72 + k] = (unsigned short)(bR[p].x & 0xFFFFu);
        Bsb[(n + 1) * 72 + k] = (unsigned short)(bR[p].x >> 16);
        Bsb[(n + 2) * 72 + k] = (unsigned short)(bR[p].y & 0xFFFFu);
        Bsb[(n + 3) * 72 + k] = (unsigned short)(bR[p].y >> 16);
      }
    }
  };

  // ---- prologue: stage tile 0 ----
  globalLoad(0);
  stageStore(0);

  for (int k0 = 0, it = 0; k0 < K; k0 += 64, ++it) {
    const int  cur  = it & 1;
    const bool more = (k0 + 64) < K;

    if (more) globalLoad(k0 + 64);     // prefetch next tile into registers

    __syncthreads();                   // publish stores to buffer `cur`

    // ---- two 32-k sub-steps on buffer `cur`: 8 WMMAs ----
    #pragma unroll
    for (int ks = 0; ks < 64; ks += 32) {
      Frag fa[2], fb[2];
      #pragma unroll
      for (int mt = 0; mt < 2; ++mt) {
        const uint4* p4 = (const uint4*)&As[cur][(wm * 32 + mt * 16 + l16) * 72 + ks];
        fa[mt].q[0] = p4[half];          // K = ks + half*8 .. +7
        fa[mt].q[1] = p4[2 + half];      // K = ks + 16 + half*8 .. +7
      }
      #pragma unroll
      for (int nt = 0; nt < 2; ++nt) {
        const uint4* p4 = (const uint4*)&Bs[cur][(wn * 32 + nt * 16 + l16) * 72 + ks];
        fb[nt].q[0] = p4[half];          // K = ks + half*8 .. +7
        fb[nt].q[1] = p4[2 + half];      // K = ks + 16 + half*8 .. +7
      }
      #pragma unroll
      for (int mt = 0; mt < 2; ++mt)
        #pragma unroll
        for (int nt = 0; nt < 2; ++nt)
          acc[mt][nt] = __builtin_amdgcn_wmma_f32_16x16x32_bf16(
              false, fa[mt].v, false, fb[nt].v, (short)0, acc[mt][nt], false, false);
    }

    if (more) stageStore(cur ^ 1);     // fill the other buffer for next iter
  }

  // ---- epilogue: scale, bias, relu ----
  #pragma unroll
  for (int nt = 0; nt < 2; ++nt) {
    const long long col = nBase + wn * 32 + nt * 16 + l16;
    const float bv = bias ? bias[col] : 0.0f;
    #pragma unroll
    for (int mt = 0; mt < 2; ++mt) {
      const long long rw0 = mBase + wm * 32 + mt * 16 + half * 8;
      #pragma unroll
      for (int r = 0; r < 8; ++r) {
        float val = acc[mt][nt][r] * scale + bv;
        if (relu) val = fmaxf(val, 0.0f);
        c[(rw0 + r) * ldc + col] = val;
      }
    }
  }
}

// ---------------------------------------------------------------------------
// Row softmax over 512 elements (one block per row).
// ---------------------------------------------------------------------------
__global__ __launch_bounds__(256) void softmax512(float* __restrict__ p)
{
  __shared__ float red[256];
  float* r = p + (long long)blockIdx.x * 512;
  const int t = threadIdx.x;
  const float a = r[t], b = r[t + 256];
  red[t] = fmaxf(a, b);
  __syncthreads();
  for (int s = 128; s > 0; s >>= 1) { if (t < s) red[t] = fmaxf(red[t], red[t + s]); __syncthreads(); }
  const float m = red[0];
  __syncthreads();
  const float e0 = __expf(a - m), e1 = __expf(b - m);
  red[t] = e0 + e1;
  __syncthreads();
  for (int s = 128; s > 0; s >>= 1) { if (t < s) red[t] += red[t + s]; __syncthreads(); }
  const float inv = 1.0f / red[0];
  r[t]       = e0 * inv;
  r[t + 256] = e1 * inv;
}

// ---------------------------------------------------------------------------
// Average attention over heads: out[b][i][j] = mean_h attn[b][h][i][j]
// ---------------------------------------------------------------------------
__global__ __launch_bounds__(256) void avg_heads(const float* __restrict__ sc,
                                                 float* __restrict__ out)
{
  const long long idx = (long long)blockIdx.x * 256 + threadIdx.x;   // < B*S*S
  const long long b   = idx >> 18;            // /(S*S)
  const long long rem = idx & (SS2_ - 1);
  const float* base = sc + (b * H_) * SS2_ + rem;
  float s = 0.0f;
  #pragma unroll
  for (int h = 0; h < H_; ++h) s += base[(long long)h * SS2_];
  out[idx] = s * (1.0f / H_);
}

// ---------------------------------------------------------------------------
// Fused residual add + LayerNorm (biased variance), one block per token.
// x[t] = LN(x[t] + add[t]) * w + b   (in place on x, D=1024, 4 elems/thread)
// ---------------------------------------------------------------------------
__global__ __launch_bounds__(256) void add_layernorm(float* __restrict__ x,
                                                     const float* __restrict__ add,
                                                     const float* __restrict__ w,
                                                     const float* __restrict__ b)
{
  __shared__ float rs_[256], rq_[256];
  float* xr = x + (long long)blockIdx.x * D_;
  const float* ar = add + (long long)blockIdx.x * D_;
  const int t = threadIdx.x;
  float v[4], s = 0.0f, sq = 0.0f;
  #pragma unroll
  for (int i = 0; i < 4; ++i) {
    v[i] = xr[t + i * 256] + ar[t + i * 256];
    s += v[i];
    sq += v[i] * v[i];
  }
  rs_[t] = s; rq_[t] = sq;
  __syncthreads();
  for (int st = 128; st > 0; st >>= 1) {
    if (t < st) { rs_[t] += rs_[t + st]; rq_[t] += rq_[t + st]; }
    __syncthreads();
  }
  const float mean = rs_[0] * (1.0f / D_);
  const float var  = rq_[0] * (1.0f / D_) - mean * mean;
  const float rstd = rsqrtf(var + 1e-5f);
  #pragma unroll
  for (int i = 0; i < 4; ++i) {
    const int c = t + i * 256;
    xr[c] = (v[i] - mean) * rstd * w[c] + b[c];
  }
}

// ---------------------------------------------------------------------------
// Host side
// ---------------------------------------------------------------------------
static inline void launch_gemm(hipStream_t st,
    const float* A, long long aS1, long long aS0,
    const float* B, long long bS1, long long bS0,
    const float* bias,
    float* C, long long cS1, long long cS0,
    int zdiv, int lda, int ldb, int ldc,
    int M, int N, int K, int batch, float scale, int relu, int btrans)
{
  dim3 grid(N / 64, M / 128, batch), blk(256);
  gemm_bf16_wmma<<<grid, blk, 0, st>>>(A, aS1, aS0, B, bS1, bS0, bias,
                                       C, cS1, cS0, zdiv, lda, ldb, ldc, K,
                                       scale, relu, btrans);
}

extern "C" void kernel_launch(void* const* d_in, const int* in_sizes, int n_in,
                              void* d_out, int out_size, void* d_ws, size_t ws_size,
                              hipStream_t stream) {
  const float* src       = (const float*)d_in[0];
  const float* in_proj_w = (const float*)d_in[1];
  const float* in_proj_b = (const float*)d_in[2];
  const float* out_w     = (const float*)d_in[3];
  const float* out_b     = (const float*)d_in[4];
  const float* ln1_w     = (const float*)d_in[5];
  const float* ln1_b     = (const float*)d_in[6];
  const float* ln2_w     = (const float*)d_in[7];
  const float* ln2_b     = (const float*)d_in[8];
  const float* w1        = (const float*)d_in[9];
  const float* b1        = (const float*)d_in[10];
  const float* w2        = (const float*)d_in[11];
  const float* b2        = (const float*)d_in[12];

  float* x        = (float*)d_out;          // [S,B,D] region of output
  float* attn_out = x + XE_;                // [L,B,S,S] region of output

  float* ws     = (float*)d_ws;
  float* qkv    = ws;                                   // [SB, 3D]  = 12.6M f
  float* scores = qkv    + (long long)SB_ * 3 * D_;     // [BH,S,S]  = 33.6M f (reused as FFN hidden)
  float* ctx    = scores + (long long)BH_ * SS2_;       // [S,B,D]   =  4.2M f
  float* tmp    = ctx    + (long long)SB_ * D_;         // [S,B,D]   =  4.2M f

  const float SCALE = 0.125f;  // 1/sqrt(HD=64)

  // x = src
  hipMemcpyAsync(x, src, (size_t)XE_ * sizeof(float), hipMemcpyDeviceToDevice, stream);

  for (int l = 0; l < L_; ++l) {
    // 1) qkv = x @ in_proj_w[l]^T + in_proj_b[l]        [4096, 3072]
    launch_gemm(stream, x, 0, 0,
                in_proj_w + (long long)l * 3 * D_ * D_, 0, 0,
                in_proj_b + (long long)l * 3 * D_,
                qkv, 0, 0,
                1, D_, D_, 3 * D_, SB_, 3 * D_, D_, 1, 1.0f, 0, 0);

    // 2) scores[b,h] = SCALE * q @ k^T                  batch BH, [512,512], K=64
    launch_gemm(stream, qkv, 3 * D_, HD_,                     // q: off = b*3D + h*64
                qkv + D_, 3 * D_, HD_,                        // k: off = b*3D + D + h*64
                nullptr,
                scores, (long long)H_ * SS2_, SS2_,           // out: z*S*S
                H_, B_ * 3 * D_, B_ * 3 * D_, S_,
                S_, S_, HD_, BH_, SCALE, 0, 0);

    // 3) softmax over last dim
    softmax512<<<BH_ * S_, 256, 0, stream>>>(scores);

    // 4) averaged attention weights -> output slice l
    avg_heads<<<(int)((long long)B_ * SS2_ / 256), 256, 0, stream>>>(
        scores, attn_out + (long long)l * B_ * SS2_);

    // 5) ctx[b,h] = attn @ v   (B given [K=S][N=HD], n contiguous) -> [S,B,D]
    launch_gemm(stream, scores, (long long)H_ * SS2_, SS2_,
                qkv + 2 * D_, 3 * D_, HD_,                    // v: off = b*3D + 2D + h*64
                nullptr,
                ctx, (long long)D_, HD_,                      // out: off = b*D + h*64
                H_, S_, B_ * 3 * D_, B_ * D_,
                S_, HD_, S_, BH_, 1.0f, 0, 1);

    // 6) attn_out = ctx @ out_w[l]^T + out_b[l] -> tmp
    launch_gemm(stream, ctx, 0, 0,
                out_w + (long long)l * D_ * D_, 0, 0,
                out_b + (long long)l * D_,
                tmp, 0, 0,
                1, D_, D_, D_, SB_, D_, D_, 1, 1.0f, 0, 0);

    // 7) x = LN1(x + tmp)
    add_layernorm<<<SB_, 256, 0, stream>>>(x, tmp, ln1_w + (long long)l * D_,
                                           ln1_b + (long long)l * D_);

    // 8) h = relu(x @ w1[l]^T + b1[l])   (reuse scores buffer) [4096,4096]
    launch_gemm(stream, x, 0, 0,
                w1 + (long long)l * DFF_ * D_, 0, 0,
                b1 + (long long)l * DFF_,
                scores, 0, 0,
                1, D_, D_, DFF_, SB_, DFF_, D_, 1, 1.0f, 1, 0);

    // 9) ff = h @ w2[l]^T + b2[l] -> tmp  [4096,1024]
    launch_gemm(stream, scores, 0, 0,
                w2 + (long long)l * D_ * DFF_, 0, 0,
                b2 + (long long)l * D_,
                tmp, 0, 0,
                1, DFF_, DFF_, D_, SB_, D_, DFF_, 1, 1.0f, 0, 0);

    // 10) x = LN2(x + tmp)
    add_layernorm<<<SB_, 256, 0, stream>>>(x, tmp, ln2_w + (long long)l * D_,
                                           ln2_b + (long long)l * D_);
  }
}
